// Understander_75763223101874
// MI455X (gfx1250) — matrix operand order repaired and checked by reading, not compiled, not run on hardware
//
#include <hip/hip_runtime.h>
#include <hip/hip_bf16.h>

// Problem constants (match reference)
#define BB   64
#define SS   256
#define TT   128
#define EE   512
#define HH   512
#define G4H  (4*HH)
#define NBLK 4          // N-tiles per wave (register blocking)

typedef __attribute__((ext_vector_type(16))) __bf16        v16bf;
typedef __attribute__((ext_vector_type(8)))  float         v8f;
typedef __attribute__((ext_vector_type(8)))  unsigned int  v8u;

__device__ __forceinline__ unsigned int f2bf(float f) {
    unsigned int u = __builtin_bit_cast(unsigned int, f);
    return ((u + 0x7FFFu + ((u >> 16) & 1u)) >> 16) & 0xFFFFu;
}
__device__ __forceinline__ float sigm(float x) { return 1.0f / (1.0f + __expf(-x)); }

// ---------------------------------------------------------------------------
// Pack a weight matrix (Nrows x K, row-major, element [n][colOff+k]) into the
// WMMA bf16 B-fragment layout, contiguous k order (same as the A loads):
//   Bp[ ((kt*tilesN + nt)*32 + lane)*8 + d ] = bf16 pair for K = kt*32 + 16*(lane>>4) + 2d
// so each lane's fragment is 8 contiguous dwords (two b128 loads).
// ---------------------------------------------------------------------------
__global__ void pack_b_bf16(const float* __restrict__ W, long srcStride, long colOff,
                            unsigned int* __restrict__ Bp, int K, int N) {
    int idx = blockIdx.x * blockDim.x + threadIdx.x;
    int total = (K >> 5) * (N >> 4) * 256;
    if (idx >= total) return;
    int d    = idx & 7;
    int lane = (idx >> 3) & 31;
    int tile = idx >> 8;
    int tilesN = N >> 4;
    int kt = tile / tilesN, nt = tile % tilesN;
    int n = nt * 16 + (lane & 15);
    int k = kt * 32 + 16 * (lane >> 4) + 2 * d;
    float f0 = W[(size_t)n * srcStride + colOff + k];
    float f1 = W[(size_t)n * srcStride + colOff + k + 1];
    Bp[idx] = f2bf(f0) | (f2bf(f1) << 16);
}

// ---------------------------------------------------------------------------
// bf16 WMMA GEMM:  C[M x N](f32) = A[M x K](bf16 rowmajor) * Bpacked
//                    (+ addend[row*addStride + n]) (+ bias[n])
// One wave computes a 16 x 64 strip: 4 independent accumulators -> 4
// back-to-back independent v_wmma per K-step; A fragment reused 4x.
// Branchless body: compile-time shapes, clamped prefetch address.
// ---------------------------------------------------------------------------
template<int N, int K, bool HAS_ADD, bool HAS_BIAS>
__global__ __launch_bounds__(256) void wmma_gemm_bf16(
    const unsigned short* __restrict__ A,
    const unsigned int*  __restrict__ Bp,
    const float* __restrict__ addend, long addStride,
    const float* __restrict__ bias,
    float* __restrict__ C, int M)
{
    constexpr int tilesN  = N >> 4;
    constexpr int ktiles  = K >> 5;
    const int lane = threadIdx.x & 31;
    const int wave = (blockIdx.x * blockDim.x + threadIdx.x) >> 5;
    constexpr int ngroups = tilesN / NBLK;
    const int mt  = wave / ngroups;
    const int nt0 = (wave % ngroups) * NBLK;
    if (mt * 16 >= M) return;

    const int lh = lane >> 4;          // lane half (0/1)
    const int ln = lane & 15;
    const int m  = mt * 16 + ln;       // A-fragment row for this lane

    const unsigned int* arowBase =
        (const unsigned int*)(A + (size_t)m * K);
    const unsigned int* bpBase =
        Bp + ((size_t)nt0 * 32 + lane) * 8;

    v8f acc[NBLK] = {};
#pragma unroll 4
    for (int kt = 0; kt < ktiles; ++kt) {
        // ---- A fragment: 16x32 bf16, contiguous k order (8 dwords per lane) ----
        const unsigned int* arow = arowBase + kt * 16;
        v8u au;
#pragma unroll
        for (int v = 0; v < 8; ++v) au[v] = arow[8 * lh + v];
        v16bf af = __builtin_bit_cast(v16bf, au);

        // Branchless prefetch of next K-tile's B group (clamped -> always valid)
        const int pkt = (kt + 1 < ktiles) ? (kt + 1) : kt;
        __builtin_prefetch(bpBase + (size_t)pkt * tilesN * 256, 0, 3);

        // ---- 4 B fragments -> 4 independent WMMAs ----
        const unsigned int* bp = bpBase + (size_t)kt * tilesN * 256;
#pragma unroll
        for (int j = 0; j < NBLK; ++j) {
            v8u bu;
#pragma unroll
            for (int v = 0; v < 8; ++v) bu[v] = bp[j * 256 + v];
            v16bf bf = __builtin_bit_cast(v16bf, bu);
            acc[j] = __builtin_amdgcn_wmma_f32_16x16x32_bf16(
                false, af, false, bf, (short)0, acc[j], false, false);
        }
    }

#pragma unroll
    for (int j = 0; j < NBLK; ++j) {
        const int n = (nt0 + j) * 16 + ln;
#pragma unroll
        for (int v = 0; v < 8; ++v) {
            int row = mt * 16 + v + 8 * lh;          // C/D VGPR layout
            float out = acc[j][v];
            if (HAS_ADD)  out += addend[(size_t)row * addStride + n];
            if (HAS_BIAS) out += bias[n];
            C[(size_t)row * N + n] = out;
        }
    }
}

// ---------------------------------------------------------------------------
// Embedding gather + bf16 cast:  Xbf[(b*S+s)*E + e] = bf16(emb[state[b,s], e])
// ---------------------------------------------------------------------------
__global__ void embed_cast(const int* __restrict__ state, const float* __restrict__ emb,
                           unsigned short* __restrict__ Xbf, int BSrows, int E) {
    int idx = blockIdx.x * blockDim.x + threadIdx.x;
    if (idx >= BSrows * E) return;
    int row = idx / E, e = idx - row * E;
    Xbf[idx] = (unsigned short)f2bf(emb[(size_t)state[row] * E + e]);
}

// bias sums + initial h/c broadcast
__global__ void prep_bias(const float* eb_ih, const float* eb_hh,
                          const float* db_ih, const float* db_hh,
                          float* ebias, float* gbias) {
    int i = blockIdx.x * blockDim.x + threadIdx.x;
    if (i >= G4H) return;
    ebias[i] = eb_ih[i] + eb_hh[i];
    gbias[i] = db_ih[i] + db_hh[i];   // decoder input is all-zero -> W_ih term vanishes
}
__global__ void init_hc(const float* h0, const float* c0,
                        float* h, float* c, unsigned short* h_bf) {
    int i = blockIdx.x * blockDim.x + threadIdx.x;
    if (i >= BB * HH) return;
    int u = i % HH;
    float hv = h0[u];
    h[i] = hv; c[i] = c0[u];
    h_bf[i] = (unsigned short)f2bf(hv);
}

// ---------------------------------------------------------------------------
// LSTM cell updates
// ---------------------------------------------------------------------------
__global__ void lstm_cell_enc(const float* __restrict__ g, float* __restrict__ h,
                              float* __restrict__ c, unsigned short* __restrict__ h_bf,
                              unsigned short* __restrict__ enc_outs_bf, int t) {
    int i = blockIdx.x * blockDim.x + threadIdx.x;
    if (i >= BB * HH) return;
    int b = i / HH, u = i - b * HH;
    const float* gb = g + (size_t)b * G4H;
    float gi = gb[u], gf = gb[HH + u], gg = gb[2 * HH + u], go = gb[3 * HH + u];
    float cn = sigm(gf) * c[i] + sigm(gi) * tanhf(gg);
    float hn = sigm(go) * tanhf(cn);
    c[i] = cn; h[i] = hn;
    unsigned short hb = (unsigned short)f2bf(hn);
    h_bf[i] = hb;
    enc_outs_bf[((size_t)b * SS + t) * HH + u] = hb;
}

__global__ void lstm_cell_dec(const float* __restrict__ g, float* __restrict__ h,
                              float* __restrict__ c, unsigned short* __restrict__ h_bf,
                              float* __restrict__ dec_states, int t) {
    int i = blockIdx.x * blockDim.x + threadIdx.x;
    if (i >= BB * HH) return;
    int b = i / HH, u = i - b * HH;
    const float* gb = g + (size_t)b * G4H;
    float gi = gb[u], gf = gb[HH + u], gg = gb[2 * HH + u], go = gb[3 * HH + u];
    float cn = sigm(gf) * c[i] + sigm(gi) * tanhf(gg);
    float hn = sigm(go) * tanhf(cn);
    c[i] = cn; h[i] = hn;
    h_bf[i] = (unsigned short)f2bf(hn);
    dec_states[((size_t)b * TT + t) * HH + u] = hn;   // output region 2
}

// ---------------------------------------------------------------------------
// score[b,t,s] = sum_h relu(enc_part[b,s,h] + d[b,h]) * W2[h] + b2
// One wave per (b,s); enc_part (33.5 MB) stays L2-resident across 128 steps.
// ---------------------------------------------------------------------------
__global__ __launch_bounds__(256) void attn_score(
    const float* __restrict__ enc_part, const float* __restrict__ dvec,
    const float* __restrict__ W2, const float* __restrict__ b2,
    float* __restrict__ scores, int t)
{
    int wave = (blockIdx.x * blockDim.x + threadIdx.x) >> 5;
    int lane = threadIdx.x & 31;
    if (wave >= BB * SS) return;
    int b = wave / SS, s = wave - b * SS;
    const float* ep = enc_part + ((size_t)b * SS + s) * HH;
    const float* dv = dvec + (size_t)b * HH;
    float acc = 0.f;
#pragma unroll
    for (int hh = lane; hh < HH; hh += 32) {
        float v = ep[hh] + dv[hh];
        acc = fmaf(fmaxf(v, 0.f), W2[hh], acc);
    }
#pragma unroll
    for (int off = 16; off; off >>= 1) acc += __shfl_down(acc, off, 32);
    if (lane == 0)
        scores[((size_t)b * TT + t) * SS + s] = acc + b2[0];
}

// ---------------------------------------------------------------------------
// Masked log-softmax over s (S == 256 == blockDim.x), one block per (b,t).
// ---------------------------------------------------------------------------
__global__ __launch_bounds__(256) void log_softmax_mask(
    const float* __restrict__ scores, const unsigned char* __restrict__ mask,
    float* __restrict__ out)
{
    int row = blockIdx.x;            // b*T + t
    int b = row / TT;
    int s = threadIdx.x;
    float x = scores[(size_t)row * SS + s];
    bool m = mask[(size_t)b * SS + s] != 0;
    float xv = m ? x : -__builtin_inff();
    __shared__ float red[256];
    red[s] = xv; __syncthreads();
    for (int off = 128; off; off >>= 1) {
        if (s < off) red[s] = fmaxf(red[s], red[s + off]);
        __syncthreads();
    }
    float mx = red[0]; __syncthreads();
    float e = (xv == -__builtin_inff()) ? 0.f : __expf(xv - mx);
    red[s] = e; __syncthreads();
    for (int off = 128; off; off >>= 1) {
        if (s < off) red[s] += red[s + off];
        __syncthreads();
    }
    out[(size_t)row * SS + s] = xv - mx - __logf(red[0]);
}

// ---------------------------------------------------------------------------
// Orchestration
// ---------------------------------------------------------------------------
static inline char* bump(char*& p, size_t bytes) {
    char* r = p;
    p += (bytes + 255) & ~(size_t)255;
    return r;
}

extern "C" void kernel_launch(void* const* d_in, const int* in_sizes, int n_in,
                              void* d_out, int out_size, void* d_ws, size_t ws_size,
                              hipStream_t stream) {
    (void)in_sizes; (void)n_in; (void)out_size; (void)ws_size;
    const int*   state  = (const int*)d_in[0];
    const unsigned char* vmask = (const unsigned char*)d_in[1];   // bool mask
    const float* emb    = (const float*)d_in[3];
    const float* eWih   = (const float*)d_in[4];
    const float* eWhh   = (const float*)d_in[5];
    const float* eBih   = (const float*)d_in[6];
    const float* eBhh   = (const float*)d_in[7];
    const float* h0     = (const float*)d_in[8];
    const float* c0     = (const float*)d_in[9];
    const float* dWhh   = (const float*)d_in[11];
    const float* dBih   = (const float*)d_in[12];
    const float* dBhh   = (const float*)d_in[13];
    const float* W1     = (const float*)d_in[14];
    const float* b1     = (const float*)d_in[15];
    const float* W2     = (const float*)d_in[16];
    const float* b2     = (const float*)d_in[17];

    float* logits_out = (float*)d_out;                                   // (B,T,S)
    float* dec_states = (float*)d_out + (size_t)BB * TT * SS;            // (B,T,H)

    // Workspace carve-up (~220 MB)
    char* p = (char*)d_ws;
    unsigned short* Xbf      = (unsigned short*)bump(p, (size_t)BB*SS*EE*2);
    unsigned int*   packWih  = (unsigned int*)  bump(p, (size_t)(EE/32)*(G4H/16)*256*4);
    unsigned int*   packWhhE = (unsigned int*)  bump(p, (size_t)(HH/32)*(G4H/16)*256*4);
    unsigned int*   packWhhD = (unsigned int*)  bump(p, (size_t)(HH/32)*(G4H/16)*256*4);
    unsigned int*   packW1e  = (unsigned int*)  bump(p, (size_t)(HH/32)*(HH/16)*256*4);
    unsigned int*   packW1d  = (unsigned int*)  bump(p, (size_t)(HH/32)*(HH/16)*256*4);
    float*          Xp       = (float*)         bump(p, (size_t)BB*SS*G4H*4);
    unsigned short* encO_bf  = (unsigned short*)bump(p, (size_t)BB*SS*HH*2);
    float*          enc_part = (float*)         bump(p, (size_t)BB*SS*HH*4);
    float*          gbuf     = (float*)         bump(p, (size_t)BB*G4H*4);
    float*          hbuf     = (float*)         bump(p, (size_t)BB*HH*4);
    float*          cbuf     = (float*)         bump(p, (size_t)BB*HH*4);
    unsigned short* h_bf     = (unsigned short*)bump(p, (size_t)BB*HH*2);
    float*          dvec     = (float*)         bump(p, (size_t)BB*HH*4);
    float*          ebias    = (float*)         bump(p, (size_t)G4H*4);
    float*          gbias    = (float*)         bump(p, (size_t)G4H*4);
    float*          scores   = (float*)         bump(p, (size_t)BB*TT*SS*4);

    // ---- one-time prep ----
    {
        int tot = BB * SS * EE;
        embed_cast<<<(tot + 255) / 256, 256, 0, stream>>>(state, emb, Xbf, BB * SS, EE);
    }
    {   // pack weights into WMMA B-fragment layout
        int nIH = (EE/32)*(G4H/16)*256;
        pack_b_bf16<<<(nIH+255)/256, 256, 0, stream>>>(eWih, EE, 0, packWih, EE, G4H);
        int nHH = (HH/32)*(G4H/16)*256;
        pack_b_bf16<<<(nHH+255)/256, 256, 0, stream>>>(eWhh, HH, 0, packWhhE, HH, G4H);
        pack_b_bf16<<<(nHH+255)/256, 256, 0, stream>>>(dWhh, HH, 0, packWhhD, HH, G4H);
        int nW1 = (HH/32)*(HH/16)*256;
        pack_b_bf16<<<(nW1+255)/256, 256, 0, stream>>>(W1, HH + HH, 0,  packW1e, HH, HH); // W1[:, :K]
        pack_b_bf16<<<(nW1+255)/256, 256, 0, stream>>>(W1, HH + HH, HH, packW1d, HH, HH); // W1[:, K:]
    }
    prep_bias<<<(G4H + 255) / 256, 256, 0, stream>>>(eBih, eBhh, dBih, dBhh, ebias, gbias);
    init_hc<<<(BB * HH + 255) / 256, 256, 0, stream>>>(h0, c0, hbuf, cbuf, h_bf);

    // ---- all-timestep input projection: Xp = Xbf @ W_ih^T + (b_ih+b_hh) ----
    {
        int waves = (BB * SS / 16) * (G4H / 16) / NBLK;       // 32768
        wmma_gemm_bf16<G4H, EE, false, true><<<waves / 8, 256, 0, stream>>>(
            Xbf, packWih, nullptr, 0, ebias, Xp, BB * SS);
    }

    // ---- encoder recurrence (sequential over S) ----
    for (int t = 0; t < SS; ++t) {
        int waves = (BB / 16) * (G4H / 16) / NBLK;            // 128
        wmma_gemm_bf16<G4H, HH, true, false><<<waves / 8, 256, 0, stream>>>(
            h_bf, packWhhE, Xp + (size_t)t * G4H, (long)SS * G4H, nullptr,
            gbuf, BB);
        lstm_cell_enc<<<(BB * HH) / 256, 256, 0, stream>>>(gbuf, hbuf, cbuf, h_bf, encO_bf, t);
    }

    // ---- attention encoder projection: enc_part = enc_outs @ W1_enc^T + b1 ----
    {
        int waves = (BB * SS / 16) * (HH / 16) / NBLK;        // 8192
        wmma_gemm_bf16<HH, HH, false, true><<<waves / 8, 256, 0, stream>>>(
            encO_bf, packW1e, nullptr, 0, b1, enc_part, BB * SS);
    }

    // ---- decoder recurrence (sequential over T) ----
    for (int t = 0; t < TT; ++t) {
        int waves = (BB / 16) * (G4H / 16) / NBLK;            // 128
        wmma_gemm_bf16<G4H, HH, false, true><<<waves / 8, 256, 0, stream>>>(
            h_bf, packWhhD, nullptr, 0, gbias, gbuf, BB);
        lstm_cell_dec<<<(BB * HH) / 256, 256, 0, stream>>>(gbuf, hbuf, cbuf, h_bf, dec_states, t);

        int waves2 = (BB / 16) * (HH / 16) / NBLK;            // 32
        wmma_gemm_bf16<HH, HH, false, false><<<waves2 / 8, 256, 0, stream>>>(
            h_bf, packW1d, nullptr, 0, nullptr, dvec, BB);

        attn_score<<<(BB * SS) / 8, 256, 0, stream>>>(enc_part, dvec, W2, b2, scores, t);
    }

    // ---- masked log-softmax into output ----
    log_softmax_mask<<<BB * TT, 256, 0, stream>>>(scores, vmask, logits_out);
}
